// LSTM_HEAD_28862180229680
// MI455X (gfx1250) — matrix operand order repaired and checked by reading, not compiled, not run on hardware
//
#include <hip/hip_runtime.h>
#include <hip/hip_bf16.h>

typedef __attribute__((ext_vector_type(16))) __bf16 v16bf;
typedef __attribute__((ext_vector_type(8)))  float  v8f;
typedef unsigned short ushort_t;

#define BS    1024
#define SEQ   168
#define CTX   64
#define HID   512
#define NGATE 2048   // 4*HID
#define KDIM  576    // 64 (x) + 512 (h) packed columns of Wt
#define TWIN  96
#define NOUT  167    // SEQ-1 predictions per batch row

// ---------- helpers ----------
__device__ __forceinline__ ushort_t f2bf(float f) {
    unsigned u = __float_as_uint(f);
    unsigned r = u + 0x7FFFu + ((u >> 16) & 1u);   // round-to-nearest-even
    return (ushort_t)(r >> 16);
}

union ABFrag { uint4 u[2]; v16bf v; };

// ---------- prep kernels ----------

// Wt[n][k] = bf16(W[src(k)][n]); src(k): k<64 -> k (x rows), k>=64 -> k+2 (h rows 66..577)
__global__ void k_prep_w(const float* __restrict__ W, ushort_t* __restrict__ Wt) {
    int idx = blockIdx.x * 256 + threadIdx.x;
    if (idx >= NGATE * KDIM) return;
    int n = idx / KDIM, k = idx % KDIM;
    int src = (k < CTX) ? k : (k + 2);
    Wt[(size_t)n * KDIM + k] = f2bf(W[(size_t)src * NGATE + n]);
}

__global__ void k_prep_x(const float* __restrict__ x, ushort_t* __restrict__ x16) {
    int idx = blockIdx.x * 256 + threadIdx.x;
    if (idx >= BS * SEQ * CTX) return;
    x16[idx] = f2bf(x[idx]);
}

// zero both h ping-pong buffers + c; seed predictions with bo
__global__ void k_init(ushort_t* __restrict__ h16, float* __restrict__ c,
                       float* __restrict__ pred, const float* __restrict__ bo) {
    int idx = blockIdx.x * 256 + threadIdx.x;
    if (idx < 2 * BS * HID) h16[idx] = 0;
    if (idx < BS * HID)     c[idx] = 0.0f;
    if (idx < BS * NOUT)    pred[idx] = bo[0];
}

// ---------- fused LSTM step: gate GEMM (bf16 WMMA) + cell + head ----------
__global__ __launch_bounds__(32) void k_lstm_step(
    const ushort_t* __restrict__ x16,   // [BS][SEQ][CTX] bf16
    const ushort_t* __restrict__ Wt,    // [NGATE][KDIM] bf16, B-operand layout
    const float*    __restrict__ yflow, // [BS][SEQ]
    const float*    __restrict__ W,     // fp32 original (rows 64,65 used)
    const float*    __restrict__ bias,  // [NGATE]
    const float*    __restrict__ Wo,    // [HID]
    float*          __restrict__ pred,  // [BS][NOUT]
    const ushort_t* __restrict__ hin,   // [BS][HID] bf16
    ushort_t*       __restrict__ hout,  // [BS][HID] bf16
    float*          __restrict__ c,     // [BS][HID]
    int t, int autoreg, float hv)
{
    const int lane = threadIdx.x;
    const int m  = lane & 15;      // A row / B col / D col within tile
    const int hi = lane >> 4;      // half-wave selector for K striping
    const int b0 = blockIdx.x << 4;
    const int j0 = blockIdx.y << 4;

    v8f acc0 = {}, acc1 = {}, acc2 = {}, acc3 = {};

    const ushort_t* bp0 = Wt + (size_t)(0 * HID + j0 + m) * KDIM;
    const ushort_t* bp1 = Wt + (size_t)(1 * HID + j0 + m) * KDIM;
    const ushort_t* bp2 = Wt + (size_t)(2 * HID + j0 + m) * KDIM;
    const ushort_t* bp3 = Wt + (size_t)(3 * HID + j0 + m) * KDIM;

    // ---- x-part of K reduction: Wt columns [0,64)
    const ushort_t* arow = x16 + ((size_t)(b0 + m) * SEQ + t) * CTX;
    #pragma unroll
    for (int k = 0; k < CTX; k += 32) {
        ABFrag a;
        const ushort_t* pa = arow + k + hi * 16;
        a.u[0] = *(const uint4*)(pa);
        a.u[1] = *(const uint4*)(pa + 8);
        const int kc = k + hi * 16;
        ABFrag w0, w1, w2, w3;
        w0.u[0] = *(const uint4*)(bp0 + kc); w0.u[1] = *(const uint4*)(bp0 + kc + 8);
        w1.u[0] = *(const uint4*)(bp1 + kc); w1.u[1] = *(const uint4*)(bp1 + kc + 8);
        w2.u[0] = *(const uint4*)(bp2 + kc); w2.u[1] = *(const uint4*)(bp2 + kc + 8);
        w3.u[0] = *(const uint4*)(bp3 + kc); w3.u[1] = *(const uint4*)(bp3 + kc + 8);
        acc0 = __builtin_amdgcn_wmma_f32_16x16x32_bf16(false, a.v, false, w0.v, (short)0, acc0, false, false);
        acc1 = __builtin_amdgcn_wmma_f32_16x16x32_bf16(false, a.v, false, w1.v, (short)0, acc1, false, false);
        acc2 = __builtin_amdgcn_wmma_f32_16x16x32_bf16(false, a.v, false, w2.v, (short)0, acc2, false, false);
        acc3 = __builtin_amdgcn_wmma_f32_16x16x32_bf16(false, a.v, false, w3.v, (short)0, acc3, false, false);
    }

    // ---- recurrent part of K reduction: Wt columns [64, 576)
    const ushort_t* hrow = hin + (size_t)(b0 + m) * HID;
    #pragma unroll 4
    for (int k = 0; k < HID; k += 32) {
        ABFrag a;
        const ushort_t* pa = hrow + k + hi * 16;
        a.u[0] = *(const uint4*)(pa);
        a.u[1] = *(const uint4*)(pa + 8);
        const int kc = CTX + k + hi * 16;
        ABFrag w0, w1, w2, w3;
        w0.u[0] = *(const uint4*)(bp0 + kc); w0.u[1] = *(const uint4*)(bp0 + kc + 8);
        w1.u[0] = *(const uint4*)(bp1 + kc); w1.u[1] = *(const uint4*)(bp1 + kc + 8);
        w2.u[0] = *(const uint4*)(bp2 + kc); w2.u[1] = *(const uint4*)(bp2 + kc + 8);
        w3.u[0] = *(const uint4*)(bp3 + kc); w3.u[1] = *(const uint4*)(bp3 + kc + 8);
        acc0 = __builtin_amdgcn_wmma_f32_16x16x32_bf16(false, a.v, false, w0.v, (short)0, acc0, false, false);
        acc1 = __builtin_amdgcn_wmma_f32_16x16x32_bf16(false, a.v, false, w1.v, (short)0, acc1, false, false);
        acc2 = __builtin_amdgcn_wmma_f32_16x16x32_bf16(false, a.v, false, w2.v, (short)0, acc2, false, false);
        acc3 = __builtin_amdgcn_wmma_f32_16x16x32_bf16(false, a.v, false, w3.v, (short)0, acc3, false, false);
    }

    // ---- epilogue: scalar rank-1 terms (y, h_since, bias), activations, cell update, head
    const int col     = j0 + m;           // hidden unit index (D-tile column = lane&15)
    const int rowbase = b0 + hi * 8;      // D-tile rows for this half-wave

    const float bi  = bias[0 * HID + col], bff = bias[1 * HID + col];
    const float bg  = bias[2 * HID + col], boo = bias[3 * HID + col];
    const float wyi = W[(size_t)64 * NGATE + 0 * HID + col];
    const float wyf = W[(size_t)64 * NGATE + 1 * HID + col];
    const float wyg = W[(size_t)64 * NGATE + 2 * HID + col];
    const float wyo = W[(size_t)64 * NGATE + 3 * HID + col];
    const float addi = hv * W[(size_t)65 * NGATE + 0 * HID + col] + bi;
    const float addf = hv * W[(size_t)65 * NGATE + 1 * HID + col] + bff;
    const float addg = hv * W[(size_t)65 * NGATE + 2 * HID + col] + bg;
    const float addo = hv * W[(size_t)65 * NGATE + 3 * HID + col] + boo;
    const float wo = Wo[col];

    #pragma unroll
    for (int r = 0; r < 8; ++r) {
        const int row = rowbase + r;
        const float yv = autoreg ? pred[(size_t)row * NOUT + (t - 1)]
                                 : yflow[(size_t)row * SEQ + t];
        float gi = acc0[r] + yv * wyi + addi;
        float gf = acc1[r] + yv * wyf + addf;
        float gg = acc2[r] + yv * wyg + addg;
        float go = acc3[r] + yv * wyo + addo;
        float iv = 1.0f / (1.0f + __expf(-gi));
        float fv = 1.0f / (1.0f + __expf(-gf));
        float ov = 1.0f / (1.0f + __expf(-go));
        float gv = tanhf(gg);
        const size_t ci = (size_t)row * HID + col;
        float cn = fv * c[ci] + iv * gv;
        c[ci] = cn;
        float hn = ov * tanhf(cn);
        hout[ci] = f2bf(hn);
        // head partial: reduce h*Wo over the 16 hidden cols this half-wave owns
        float part = hn * wo;
        part += __shfl_xor(part, 1, 32);
        part += __shfl_xor(part, 2, 32);
        part += __shfl_xor(part, 4, 32);
        part += __shfl_xor(part, 8, 32);
        if (m == 0) atomicAdd(&pred[(size_t)row * NOUT + t], part);
    }
}

// ---------- host ----------
extern "C" void kernel_launch(void* const* d_in, const int* in_sizes, int n_in,
                              void* d_out, int out_size, void* d_ws, size_t ws_size,
                              hipStream_t stream) {
    (void)in_sizes; (void)n_in; (void)out_size; (void)ws_size;
    const float* x_dyn  = (const float*)d_in[0];
    const float* y_flow = (const float*)d_in[1];
    const float* W      = (const float*)d_in[2];
    const float* b      = (const float*)d_in[3];
    const float* Wo     = (const float*)d_in[4];
    const float* bo     = (const float*)d_in[5];
    float* pred = (float*)d_out;

    char* ws = (char*)d_ws;
    ushort_t* Wt  = (ushort_t*)(ws);                              // 2048*576*2    = 2359296 B
    ushort_t* x16 = (ushort_t*)(ws + 2359296);                    // 1024*168*64*2 = 22020096 B
    ushort_t* h16 = (ushort_t*)(ws + 2359296 + 22020096);         // 2*1024*512*2  = 2097152 B
    float*    c   = (float*)   (ws + 2359296 + 22020096 + 2097152); // 1024*512*4  = 2097152 B

    k_prep_w<<<(NGATE * KDIM + 255) / 256, 256, 0, stream>>>(W, Wt);
    k_prep_x<<<(BS * SEQ * CTX + 255) / 256, 256, 0, stream>>>(x_dyn, x16);
    k_init  <<<(2 * BS * HID + 255) / 256, 256, 0, stream>>>(h16, c, pred, bo);

    const dim3 grid(BS / 16, HID / 16, 1);
    for (int t = 0; t < NOUT; ++t) {
        const int autoreg = (t >= TWIN) ? 1 : 0;
        const float hv = autoreg ? (float)(t - TWIN + 1) / 24.0f : 0.0f;
        ushort_t* hin  = h16 + (size_t)(t & 1) * BS * HID;
        ushort_t* hout = h16 + (size_t)((t + 1) & 1) * BS * HID;
        k_lstm_step<<<grid, 32, 0, stream>>>(x16, Wt, y_flow, W, b, Wo,
                                             pred, hin, hout, c, t, autoreg, hv);
    }
}
